// ModConv_4801773437058
// MI455X (gfx1250) — matrix operand order repaired and checked by reading, not compiled, not run on hardware
//
#include <hip/hip_runtime.h>

// ---------------------------------------------------------------------------
// StyleGAN2 modulated conv2d for MI455X (gfx1250), implicit-GEMM via bf16 WMMA
//   b=16, c_in=c_out=256, h=w=64, k=3  ->  GEMM per sample: M=4096 N=256 K=2304
// Round 2: double-buffered LDS staging via global_load_async_to_lds_b128
// (ASYNCcnt path) + A-fragment prefetch pipelining.
// ---------------------------------------------------------------------------

#define B_      16
#define CIN     256
#define COUT    256
#define H_      64
#define W_      64
#define HP      66          // padded spatial (pad=1 each side)
#define KSTEPS  72          // K=2304 = 72 steps of 32, K order = (tap r)*256 + c_in

typedef __bf16 v16bf __attribute__((ext_vector_type(16)));
typedef float  v8f   __attribute__((ext_vector_type(8)));
typedef float  fvec4 __attribute__((ext_vector_type(4)));

union Frag16 { v16bf v; fvec4 f4[2]; };

// workspace layout (bytes)
static constexpr size_t S_BYTES   = (size_t)B_ * CIN * 4;                    // 16 KB
static constexpr size_t BWS_BYTES = (size_t)B_ * KSTEPS * 16 * 32 * 16 * 2;  // 18.87 MB
static constexpr size_t XB_BYTES  = (size_t)B_ * HP * HP * CIN * 2;          // 35.68 MB

// ---------------------------------------------------------------------------
// Kernel 1: s[b][c_in] = dot(w_style[b], aff_w[c_in]) + aff_b[c_in] + 1
// ---------------------------------------------------------------------------
__global__ void style_kernel(const float* __restrict__ wsty,
                             const float* __restrict__ aff_w,
                             const float* __restrict__ aff_b,
                             float* __restrict__ s) {
  __shared__ float wloc[512];
  const int b = blockIdx.x, t = threadIdx.x;
  wloc[t]       = wsty[b * 512 + t];
  wloc[t + 256] = wsty[b * 512 + t + 256];
  __syncthreads();
  float acc = aff_b[t] + 1.0f;
  const float* row = aff_w + (size_t)t * 512;
  #pragma unroll 8
  for (int i = 0; i < 512; ++i) acc += wloc[i] * row[i];
  s[b * CIN + t] = acc;
}

// ---------------------------------------------------------------------------
// Kernel 2: x NCHW f32 -> padded NHWC bf16 (xb[b][y][x][c]), via LDS transpose
// grid (64 y, 16 b), 256 threads. Borders pre-zeroed with hipMemsetAsync.
// ---------------------------------------------------------------------------
__global__ void pad_kernel(const float* __restrict__ x, __bf16* __restrict__ xb) {
  __shared__ float tile[64][65];
  const int y = blockIdx.x, b = blockIdx.y, t = threadIdx.x;
  const int q = t & 3;
  for (int c0 = 0; c0 < CIN; c0 += 64) {
    __syncthreads();
    {   // coalesced read: 4 threads cover one 64-float channel row
      const int c = c0 + (t >> 2);
      const float* src = x + (((size_t)b * CIN + c) * H_ + y) * W_ + q * 16;
      #pragma unroll
      for (int i = 0; i < 16; ++i) tile[c - c0][q * 16 + i] = src[i];
    }
    __syncthreads();
    {   // coalesced write: 16 consecutive bf16 channels per thread
      const int xx = t >> 2;
      __bf16* dst = xb + (((size_t)b * HP + (y + 1)) * HP + (xx + 1)) * CIN + c0 + q * 16;
      #pragma unroll
      for (int j = 0; j < 16; ++j) dst[j] = (__bf16)tile[q * 16 + j][xx];
    }
  }
}

// ---------------------------------------------------------------------------
// Kernel 3: modulate + demodulate, emit bf16 weights pre-swizzled into WMMA
// B-fragment order: Bws[b][kt][nt][lane][i]  (lane<16: K rows 0..15 of the
// 32-K step, lane>=16: K rows 16..31; N = lane&15; i = K&15).
// grid (256 c_out, 16 b), 256 threads (t = c_in).
// ---------------------------------------------------------------------------
__global__ void modw_kernel(const float* __restrict__ W,
                            const float* __restrict__ s,
                            __bf16* __restrict__ Bws) {
  const int cout = blockIdx.x, b = blockIdx.y, t = threadIdx.x;   // t = c_in
  const float sv = s[b * CIN + t];
  const float* wrow = W + ((size_t)cout * CIN + t) * 9;
  float w9[9], ssq = 0.0f;
  #pragma unroll
  for (int r = 0; r < 9; ++r) { w9[r] = wrow[r] * sv; ssq += w9[r] * w9[r]; }

  __shared__ float red[256];
  red[t] = ssq; __syncthreads();
  for (int o = 128; o > 0; o >>= 1) { if (t < o) red[t] += red[t + o]; __syncthreads(); }
  const float dmod = rsqrtf(red[0] + 1e-8f);

  const int klocal = t & 31, i = klocal & 15, half = klocal >> 4;
  const int lane = half * 16 + (cout & 15);
  const int nt = cout >> 4;
  const int ktbase = t >> 5;                 // c_in block within tap (0..7)
  #pragma unroll
  for (int r = 0; r < 9; ++r) {
    const int kt = r * 8 + ktbase;           // K index j = r*256 + c_in -> step
    const size_t idx = ((((size_t)(b * KSTEPS + kt) * 16 + nt) * 32 + lane) * 16 + i);
    Bws[idx] = (__bf16)(w9[r] * dmod);
  }
}

// ---------------------------------------------------------------------------
// Kernel 4: implicit-GEMM conv via v_wmma_f32_16x16x32_bf16.
// grid (16 pixel-groups of 4 rows, 16 b), 256 threads (8 waves).
// Wave tile: 32 pixels x 256 c_out -> 32 WMMAs per 32-wide K step, 72 steps.
// B: 16 KB K-slice staged into a 2x16KB LDS double buffer with
//    global_load_async_to_lds_b128 (ASYNCcnt), overlapped with compute.
// A: per-lane b128 loads from padded NHWC bf16, prefetched one K-step ahead.
// ---------------------------------------------------------------------------
__global__ __launch_bounds__(256, 1)
void gemm_kernel(const __bf16* __restrict__ Bws,
                 const __bf16* __restrict__ xb,
                 float* __restrict__ out) {
  __shared__ fvec4 ldsB[2048];                      // 2 x 16 KB; reused by epilogue
  const int b = blockIdx.y, pg = blockIdx.x, t = threadIdx.x;
  const int wave = t >> 5, lane = t & 31;
  const int nlo = lane & 15, hi = lane >> 4;
  const int y = pg * 4 + (wave >> 1);               // image row of this wave
  const int xbase = (wave & 1) * 32;                // 32-pixel segment

  v8f acc[2][16] = {};                              // [m-subtile][n-subtile]

  const char* Bbase = reinterpret_cast<const char*>(Bws)
                      + (size_t)b * KSTEPS * 16384 + (size_t)t * 64;
  const __bf16* xbb = xb + (size_t)b * HP * HP * CIN;
  const unsigned ldsbase = (unsigned)(size_t)(&ldsB[0]) + (unsigned)t * 64;

  // async-stage one 16 KB B K-slice (64 B per thread) into buffer `buf`
  auto stageB = [&](int buf, int kt) {
    const unsigned loff = ldsbase + (unsigned)buf * 16384u;
    const unsigned long long g = (unsigned long long)(size_t)(Bbase + (size_t)kt * 16384);
    #pragma unroll
    for (int i = 0; i < 4; ++i)
      asm volatile("global_load_async_to_lds_b128 %0, %1, off"
                   :: "v"(loff + (unsigned)(i * 16)),
                      "v"(g + (unsigned long long)(i * 16))
                   : "memory");
  };

  // A-fragment loads for K-step kt (lane<16: K {0..15}; lane>=16: {16..31})
  auto loadA = [&](Frag16 af[2], int kt) {
    const int r = kt >> 3, kh = r / 3, kw = r - kh * 3;
    const int cb = (kt & 7) * 32;
    #pragma unroll
    for (int ms = 0; ms < 2; ++ms) {
      const int xi = xbase + ms * 16 + nlo + kw;    // padded coords (no branch)
      const int yi = y + kh;
      const fvec4* ap = reinterpret_cast<const fvec4*>(
          xbb + ((size_t)yi * HP + xi) * CIN + cb + hi * 16);
      af[ms].f4[0] = ap[0];                         // channels +0..7
      af[ms].f4[1] = ap[1];                         // channels +8..15
    }
  };

  Frag16 acur[2], anxt[2];
  stageB(0, 0);                                     // prologue: stage K-step 0
  loadA(acur, 0);

  #pragma unroll 2
  for (int kt = 0; kt < KSTEPS; ++kt) {
    const int buf = kt & 1;
    // my async writes into `buf` done; barrier makes everyone's visible.
    asm volatile("s_wait_asynccnt 0x0" ::: "memory");
    __syncthreads();
    if (kt + 1 < KSTEPS) {
      // All waves passed the barrier, i.e. finished consuming buf^1 at kt-1
      // (their ds_loads were waited before the WMMAs that fed the barrier),
      // so buf^1 is free to overwrite. Overlaps with this step's 32 WMMAs.
      stageB(buf ^ 1, kt + 1);
      loadA(anxt, kt + 1);
    }
    const fvec4* lb = &ldsB[buf * 1024];
    #pragma unroll
    for (int n = 0; n < 16; ++n) {
      Frag16 bfr;
      bfr.f4[0] = lb[n * 64 + lane * 2 + 0];        // lane-contiguous 32 B
      bfr.f4[1] = lb[n * 64 + lane * 2 + 1];
      acc[0][n] = __builtin_amdgcn_wmma_f32_16x16x32_bf16(
          false, acur[0].v, false, bfr.v, (short)0, acc[0][n], false, false);
      acc[1][n] = __builtin_amdgcn_wmma_f32_16x16x32_bf16(
          false, acur[1].v, false, bfr.v, (short)0, acc[1][n], false, false);
    }
    acur[0] = anxt[0];
    acur[1] = anxt[1];
  }

  // Epilogue: transpose each 16x16 tile through LDS (stride 17 = conflict-free)
  // so NCHW f32 stores are coalesced along x (16-lane 64 B segments).
  __syncthreads();
  float* ep = reinterpret_cast<float*>(ldsB) + wave * 272;   // 16x17 per wave
  #pragma unroll
  for (int ms = 0; ms < 2; ++ms) {
    const int x0 = xbase + ms * 16;
    #pragma unroll
    for (int n = 0; n < 16; ++n) {
      #pragma unroll
      for (int v = 0; v < 8; ++v)                   // D layout: M=v+8*hi, N=nlo
        ep[(v + 8 * hi) * 17 + nlo] = acc[ms][n][v];
      asm volatile("" ::: "memory");                // same-wave DS ops in-order
      #pragma unroll
      for (int v2 = 0; v2 < 8; ++v2) {
        const int c = v2 * 2 + hi;                  // c_out within 16-tile
        out[(((size_t)b * COUT + n * 16 + c) * H_ + y) * W_ + x0 + nlo] =
            ep[nlo * 17 + c];
      }
      asm volatile("" ::: "memory");
    }
  }
}

// ---------------------------------------------------------------------------
extern "C" void kernel_launch(void* const* d_in, const int* in_sizes, int n_in,
                              void* d_out, int out_size, void* d_ws, size_t ws_size,
                              hipStream_t stream) {
  (void)in_sizes; (void)n_in; (void)out_size; (void)ws_size;
  const float* x     = (const float*)d_in[0];   // [16,256,64,64]
  const float* wsty  = (const float*)d_in[1];   // [16,512]
  const float* W     = (const float*)d_in[2];   // [256,256,3,3]
  const float* affw  = (const float*)d_in[3];   // [256,512]
  const float* affb  = (const float*)d_in[4];   // [256]
  float* out = (float*)d_out;                   // [16,256,64,64]

  char* ws = (char*)d_ws;
  float*  s   = (float*)ws;
  __bf16* Bws = (__bf16*)(ws + S_BYTES);
  __bf16* xb  = (__bf16*)(ws + S_BYTES + BWS_BYTES);

  hipMemsetAsync(xb, 0, XB_BYTES, stream);                       // zero padding
  style_kernel<<<dim3(B_), dim3(256), 0, stream>>>(wsty, affw, affb, s);
  pad_kernel<<<dim3(H_, B_), dim3(256), 0, stream>>>(x, xb);
  modw_kernel<<<dim3(COUT, B_), dim3(256), 0, stream>>>(W, s, Bws);
  gemm_kernel<<<dim3(16, B_), dim3(256), 0, stream>>>(Bws, xb, out);
}
